// TTM_22351009808657
// MI455X (gfx1250) — matrix operand order repaired and checked by reading, not compiled, not run on hardware
//
#include <hip/hip_runtime.h>

typedef _Float16 half_t;
typedef _Float16 v16h __attribute__((ext_vector_type(16)));
typedef float    v8f  __attribute__((ext_vector_type(8)));

#define BATCH 8192
#define DIN   1024
#define DOUT  4096
#define RANK  64

#define BM 128
#define BN 128
#define BK 32
#define BKP 32            // LDS row stride (halves); 64B rows, 16B-aligned chunks
#define NT (DIN / BK)     // 32 K-steps

// CDNA5 async global->LDS copy (16B per lane), tracked by ASYNCcnt.
__device__ __forceinline__ void async_copy_b128(uint32_t lds_off, uint64_t gaddr) {
  asm volatile("global_load_async_to_lds_b128 %0, %1, off"
               :: "v"(lds_off), "v"(gaddr) : "memory");
}

// -------- Precompute 1: A[ip=128, jqs=8192] = sum_r g1[ip,r] * g2[r,jqs] ------
__global__ __launch_bounds__(256) void tt_merge12(const float* __restrict__ g1,
                                                  const float* __restrict__ g2,
                                                  float* __restrict__ Amat) {
  int idx = blockIdx.x * 256 + threadIdx.x;   // 128*8192 threads
  int ip  = idx >> 13;
  int jqs = idx & 8191;
  float acc = 0.f;
#pragma unroll
  for (int r = 0; r < RANK; ++r)
    acc += g1[ip * RANK + r] * g2[r * 8192 + jqs];
  Amat[idx] = acc;
}

// -------- Precompute 2: Wt[n=4096][kd=1024] (f16, N-major) --------------------
// n = (p*16+q)*16+m ; kd = (i*8+j)*16+k ; Wt[n][kd] = sum_s A[(i*16+p)][(j*16+q)*64+s]*g3[s,k,m]
__global__ __launch_bounds__(256) void tt_merge3(const float* __restrict__ Amat,
                                                 const float* __restrict__ g3,
                                                 half_t* __restrict__ Wt) {
  int idx = blockIdx.x * 256 + threadIdx.x;   // 4096*1024 threads
  int n  = idx >> 10;
  int kd = idx & 1023;
  int m = n & 15, q = (n >> 4) & 15, p = n >> 8;
  int k = kd & 15, j = (kd >> 4) & 7, i = kd >> 7;
  const float* Arow = Amat + (i * 16 + p) * 8192 + (j * 16 + q) * 64;
  const float* g3p  = g3 + k * 16 + m;        // stride 256 over s
  float acc = 0.f;
#pragma unroll
  for (int s = 0; s < RANK; ++s)
    acc += Arow[s] * g3p[s * 256];
  Wt[idx] = (half_t)acc;
}

// -------- Precompute 3: Xh = f16(X), enables 16B async-to-LDS staging ---------
__global__ __launch_bounds__(256) void x_cvt(const float* __restrict__ X,
                                             half_t* __restrict__ Xh) {
  int idx = (blockIdx.x * 256 + threadIdx.x) * 4;
  float4 v = *(const float4*)(X + idx);
  struct alignas(8) H4 { half_t a, b, c, d; } o;
  o.a = (half_t)v.x; o.b = (half_t)v.y; o.c = (half_t)v.z; o.d = (half_t)v.w;
  *(H4*)(Xh + idx) = o;
}

// -------- Main GEMM: Y[8192,4096] = Xh @ W, f16 WMMA, f32 acc -----------------
// Double-buffered LDS fed by global_load_async_to_lds_b128, s_wait_asynccnt sync.
union FragU { v16h v; uint4 q[2]; };

__global__ __launch_bounds__(256) void tt_gemm(const half_t* __restrict__ Xh,
                                               const half_t* __restrict__ Wt,
                                               float* __restrict__ Y) {
  __shared__ half_t Xs[2][BM * BKP];   // 2 x 8KB
  __shared__ half_t Ws[2][BN * BKP];   // 2 x 8KB  (total 32KB LDS)

  const int tid   = threadIdx.x;
  const int lane  = tid & 31;
  const int wave  = tid >> 5;
  const int waveM = wave >> 1;   // 0..3 -> 32-row slab
  const int waveN = wave & 1;    // 0..1 -> 64-col slab

  const int rowBase = blockIdx.y * BM;
  const int colBase = blockIdx.x * BN;

  // Staging map: each thread owns two 16B chunks per tile per operand.
  const int sr = tid >> 2;            // row 0..63 (and sr+64)
  const int sc = (tid & 3) << 3;      // half col 0,8,16,24

  const uint64_t xg  = (uint64_t)(uintptr_t)(Xh + (size_t)(rowBase + sr)      * DIN + sc);
  const uint64_t xg2 = (uint64_t)(uintptr_t)(Xh + (size_t)(rowBase + sr + 64) * DIN + sc);
  const uint64_t wg  = (uint64_t)(uintptr_t)(Wt + (size_t)(colBase + sr)      * DIN + sc);
  const uint64_t wg2 = (uint64_t)(uintptr_t)(Wt + (size_t)(colBase + sr + 64) * DIN + sc);

  // LDS aperture keeps the byte offset in addr[31:0] -> truncation yields LDS offset.
  uint32_t xl[2], xl2[2], wl[2], wl2[2];
#pragma unroll
  for (int b = 0; b < 2; ++b) {
    xl[b]  = (uint32_t)(uintptr_t)&Xs[b][sr * BKP + sc];
    xl2[b] = (uint32_t)(uintptr_t)&Xs[b][(sr + 64) * BKP + sc];
    wl[b]  = (uint32_t)(uintptr_t)&Ws[b][sr * BKP + sc];
    wl2[b] = (uint32_t)(uintptr_t)&Ws[b][(sr + 64) * BKP + sc];
  }

  // Prologue: stage tile 0 into buffer 0.
  async_copy_b128(xl[0],  xg);
  async_copy_b128(xl2[0], xg2);
  async_copy_b128(wl[0],  wg);
  async_copy_b128(wl2[0], wg2);

  const int lrow   = lane & 15;
  const int kbaseA = (lane < 16) ? 0 : 16;   // A: lanes 0-15 K=0..15, lanes 16-31 K=16..31
  const int kbaseB = (lane < 16) ? 0 : 16;   // B: lanes 0-15 K=0..15, lanes 16-31 K=16..31

  v8f acc[2][4];
#pragma unroll
  for (int a = 0; a < 2; ++a)
#pragma unroll
    for (int b = 0; b < 4; ++b)
      acc[a][b] = {};

  // Single-body, branch-free pipeline loop: keep accumulators in place so
  // WMMA D==C needs no inter-instruction copies or hazard NOPs.
#pragma clang loop unroll(disable)
  for (int kt = 0; kt < NT; ++kt) {
    const int cur = kt & 1;
    const int nxt = cur ^ 1;

    // Always prefetch: on the last iteration wrap to tile 0 — those dangling
    // copies land in a buffer that is never read again (S_ENDPGM waits idle).
    const int ktn = (kt + 1 == NT) ? 0 : kt + 1;
    const uint64_t d = (uint64_t)ktn * (BK * sizeof(half_t));
    async_copy_b128(xl[nxt],  xg  + d);
    async_copy_b128(xl2[nxt], xg2 + d);
    async_copy_b128(wl[nxt],  wg  + d);
    async_copy_b128(wl2[nxt], wg2 + d);
    // Oldest 4 copies in the per-wave FIFO are the current tile's: ASYNCcnt<=4.
    asm volatile("s_wait_asynccnt 0x4" ::: "memory");
    __syncthreads();   // all waves' copies for tile kt have landed

    FragU afrag[2];
#pragma unroll
    for (int fm = 0; fm < 2; ++fm) {
      const half_t* src = &Xs[cur][(waveM * 32 + fm * 16 + lrow) * BKP];
      afrag[fm].q[0] = *(const uint4*)(src + kbaseA);
      afrag[fm].q[1] = *(const uint4*)(src + kbaseA + 8);
    }
    FragU bfrag[4];
#pragma unroll
    for (int fn = 0; fn < 4; ++fn) {
      const half_t* src = &Ws[cur][(waveN * 64 + fn * 16 + lrow) * BKP];
      bfrag[fn].q[0] = *(const uint4*)(src + kbaseB);
      bfrag[fn].q[1] = *(const uint4*)(src + kbaseB + 8);
    }
#pragma unroll
    for (int fm = 0; fm < 2; ++fm)
#pragma unroll
      for (int fn = 0; fn < 4; ++fn)
        acc[fm][fn] = __builtin_amdgcn_wmma_f32_16x16x32_f16(
            false, afrag[fm].v, false, bfrag[fn].v,
            (short)0, acc[fm][fn], false, false);

    __syncthreads();   // reads of buffer cur done before it is re-filled
  }

  // D layout: lane -> col = lane%16, rows = v + (lane<16 ? 0 : 8)
  const int rbase = (lane < 16) ? 0 : 8;
#pragma unroll
  for (int fm = 0; fm < 2; ++fm)
#pragma unroll
    for (int fn = 0; fn < 4; ++fn)
#pragma unroll
      for (int v = 0; v < 8; ++v) {
        int row = rowBase + waveM * 32 + fm * 16 + rbase + v;
        int col = colBase + waveN * 64 + fn * 16 + lrow;
        Y[(size_t)row * DOUT + col] = acc[fm][fn][v];
      }
}

extern "C" void kernel_launch(void* const* d_in, const int* in_sizes, int n_in,
                              void* d_out, int out_size, void* d_ws, size_t ws_size,
                              hipStream_t stream) {
  const float* x  = (const float*)d_in[0];
  const float* g1 = (const float*)d_in[1];
  const float* g2 = (const float*)d_in[2];
  const float* g3 = (const float*)d_in[3];
  float* y = (float*)d_out;

  // Workspace layout: Amat f32 4MB | Wt f16 8MB | Xh f16 16MB  (28MB total)
  float*  Amat = (float*)d_ws;
  half_t* Wt   = (half_t*)((char*)d_ws + (size_t)4  * 1024 * 1024);
  half_t* Xh   = (half_t*)((char*)d_ws + (size_t)12 * 1024 * 1024);

  tt_merge12<<<dim3((128 * 8192) / 256),        dim3(256), 0, stream>>>(g1, g2, Amat);
  tt_merge3 <<<dim3((4096 * 1024) / 256),       dim3(256), 0, stream>>>(Amat, g3, Wt);
  x_cvt     <<<dim3((BATCH * DIN) / 4 / 256),   dim3(256), 0, stream>>>(x, Xh);
  tt_gemm   <<<dim3(DOUT / BN, BATCH / BM),     dim3(256), 0, stream>>>(Xh, Wt, y);
}
